// SCLModel_83665962926884
// MI455X (gfx1250) — compile-verified, matrix operand order checked by reading, not run on hardware
//
#include <hip/hip_runtime.h>
#include <math.h>

// CDNA5 / gfx1250 wave32 WMMA types
typedef __attribute__((ext_vector_type(16))) _Float16 v16h;
typedef __attribute__((ext_vector_type(8)))  float    v8f;

#define WAVES     4
#define BLOCK     (WAVES * 32)
#define HSTR      33     // padded row stride (floats) for h buffer  (bank spread)
#define ZSTR      129    // padded row stride (floats) for z buffer  (bank spread)
#define NEG_SLOPE 0.2f

// K index held by element e of a v16h A/B operand for lane-half g (ISA 7.12.2,
// 16-bit A 16x32: lanes 0-15 -> K {0..7,16..23}, lanes 16-31 -> K {8..15,24..31})
__device__ __forceinline__ int kmap(int e, int g) {
  return ((e < 8) ? e : (e + 8)) + 8 * g;
}

__device__ __forceinline__ v8f wmma_split3(v16h ah, v16h al, v16h bh, v16h bl, v8f c) {
  // (ah+al)*(bh+bl) ~= ah*bh + al*bh + ah*bl   (f32-ish precision from f16 WMMA)
  c = __builtin_amdgcn_wmma_f32_16x16x32_f16(false, ah, false, bh, (short)0, c, false, false);
  c = __builtin_amdgcn_wmma_f32_16x16x32_f16(false, al, false, bh, (short)0, c, false, false);
  c = __builtin_amdgcn_wmma_f32_16x16x32_f16(false, ah, false, bl, (short)0, c, false, false);
  return c;
}

// One 3-node GATv2 pass for a single sample (per-lane), accumulating the
// target-summed output (global add pool contribution) into hout[16].
__device__ __forceinline__ void gatv2_accum(
    const float x[3][14], const float pos[3][2],
    const float* sWl, const float* sbl, const float* sWr, const float* sbr,
    const float* sWe, const float* satt, const float* sbias, float hout[16])
{
  // Edge attributes (dx, dy, dist) with self-loop = mean of incoming non-self
  float attr[3][3][3];
#pragma unroll
  for (int i = 0; i < 3; ++i) {
    float a0 = 0.f, a1 = 0.f, a2 = 0.f;
#pragma unroll
    for (int j = 0; j < 3; ++j) {
      if (j == i) continue;
      float dx = pos[j][0] - pos[i][0];
      float dy = pos[j][1] - pos[i][1];
      float sq = dx * dx + dy * dy;
      float dist = (sq > 0.f) ? sqrtf(sq) : 0.f;
      attr[j][i][0] = dx; attr[j][i][1] = dy; attr[j][i][2] = dist;
      a0 += dx; a1 += dy; a2 += dist;
    }
    attr[i][i][0] = 0.5f * a0; attr[i][i][1] = 0.5f * a1; attr[i][i][2] = 0.5f * a2;
  }
  // Source transform xl = x @ Wl + bl
  float xl[3][16];
#pragma unroll
  for (int j = 0; j < 3; ++j) {
#pragma unroll
    for (int c = 0; c < 16; ++c) {
      float s = sbl[c];
#pragma unroll
      for (int k = 0; k < 14; ++k) s = fmaf(x[j][k], sWl[k * 16 + c], s);
      xl[j][c] = s;
    }
  }
#pragma unroll
  for (int i = 0; i < 3; ++i) {
    // Target transform row xr_i = x_i @ Wr + br
    float xr[16];
#pragma unroll
    for (int c = 0; c < 16; ++c) {
      float s = sbr[c];
#pragma unroll
      for (int k = 0; k < 14; ++k) s = fmaf(x[i][k], sWr[k * 16 + c], s);
      xr[c] = s;
    }
    // logits over sources j
    float lg[3];
#pragma unroll
    for (int j = 0; j < 3; ++j) {
      float s = 0.f;
#pragma unroll
      for (int c = 0; c < 16; ++c) {
        float t = xl[j][c] + xr[c];
        t = fmaf(attr[j][i][0], sWe[c],      t);
        t = fmaf(attr[j][i][1], sWe[16 + c], t);
        t = fmaf(attr[j][i][2], sWe[32 + c], t);
        t = (t > 0.f) ? t : NEG_SLOPE * t;           // leaky_relu
        s = fmaf(satt[c], t, s);
      }
      lg[j] = s;
    }
    // softmax over j
    float mx  = fmaxf(lg[0], fmaxf(lg[1], lg[2]));
    float e0 = expf(lg[0] - mx), e1 = expf(lg[1] - mx), e2 = expf(lg[2] - mx);
    float inv = 1.f / (e0 + e1 + e2);
    float al0 = e0 * inv, al1 = e1 * inv, al2 = e2 * inv;
#pragma unroll
    for (int c = 0; c < 16; ++c) {
      float o = sbias[c];
      o = fmaf(al0, xl[0][c], o);
      o = fmaf(al1, xl[1][c], o);
      o = fmaf(al2, xl[2][c], o);
      hout[c] += fmaxf(o, 0.f);                       // relu + add pool
    }
  }
}

// No occupancy floor: with the MLP loops rolled (unroll 1) the natural register
// peak is the GATv2 phase (~200 VGPRs), so the allocator lands under 256 with
// zero spill code and good occupancy on its own.
__global__ __launch_bounds__(BLOCK)
void scl_fused_kernel(
    const float* __restrict__ agent_pos, const float* __restrict__ agent_vel,
    const float* __restrict__ rel_lm,    const float* __restrict__ other_pos,
    const float* __restrict__ lm_pos,
    const float* __restrict__ Wl, const float* __restrict__ bl,
    const float* __restrict__ Wr, const float* __restrict__ br,
    const float* __restrict__ We, const float* __restrict__ att,
    const float* __restrict__ bias0,
    const float* __restrict__ W1, const float* __restrict__ b1,
    const float* __restrict__ W2, const float* __restrict__ b2,
    float* __restrict__ out)
{
  __shared__ float sWl[224], sbl[16], sWr[224], sbr[16];
  __shared__ float sWe[48], satt[16], sbias[16], sb1[128], sb2[32];
  __shared__ float hbuf[WAVES * 32 * HSTR];   // per-wave 32 samples x 32 feats (padded)
  __shared__ float zbuf[WAVES * 16 * ZSTR];   // per-wave 16 samples x 128 hidden (padded)

  // hint the MLP weights toward cache (gfx1250 global_prefetch)
  __builtin_prefetch(W1, 0, 3);
  __builtin_prefetch(W2, 0, 3);

  // ---- stage small GAT/MLP weights into LDS ----
  for (int i = threadIdx.x; i < 224; i += BLOCK) { sWl[i] = Wl[i]; sWr[i] = Wr[i]; }
  for (int i = threadIdx.x; i < 16;  i += BLOCK) {
    sbl[i] = bl[i]; sbr[i] = br[i]; satt[i] = att[i]; sbias[i] = bias0[i];
  }
  for (int i = threadIdx.x; i < 48;  i += BLOCK) sWe[i] = We[i];
  for (int i = threadIdx.x; i < 128; i += BLOCK) sb1[i] = b1[i];
  for (int i = threadIdx.x; i < 32;  i += BLOCK) sb2[i] = b2[i];
  __syncthreads();

  const int lane = threadIdx.x & 31;
  const int wave = threadIdx.x >> 5;
  const int gw   = blockIdx.x * WAVES + wave;  // global wave id
  const int s    = gw * 32 + lane;             // sample handled by this lane

  // ================= GATv2 (per-lane, one sample) =================
  float x[3][14], pos[3][2];

  // ---- agent graph ----
#pragma unroll
  for (int k = 0; k < 3; ++k) {
    pos[k][0] = agent_pos[s * 6 + k * 2 + 0];
    pos[k][1] = agent_pos[s * 6 + k * 2 + 1];
    x[k][0] = pos[k][0]; x[k][1] = pos[k][1];
    x[k][2] = agent_vel[s * 6 + k * 2 + 0];
    x[k][3] = agent_vel[s * 6 + k * 2 + 1];
#pragma unroll
    for (int d = 0; d < 6; ++d) x[k][4 + d]  = rel_lm[s * 18 + k * 6 + d];
#pragma unroll
    for (int d = 0; d < 4; ++d) x[k][10 + d] = other_pos[s * 12 + k * 4 + d];
  }
  float hA[16];
#pragma unroll
  for (int c = 0; c < 16; ++c) hA[c] = 0.f;
  gatv2_accum(x, pos, sWl, sbl, sWr, sbr, sWe, satt, sbias, hA);

  // ---- objective (landmark) graph ----
  float p[3][2];
#pragma unroll
  for (int k = 0; k < 3; ++k) {
    p[k][0] = lm_pos[s * 18 + k * 2 + 0];   // row 0 of [3][6] (rows identical)
    p[k][1] = lm_pos[s * 18 + k * 2 + 1];
  }
#pragma unroll
  for (int k = 0; k < 3; ++k) {
    pos[k][0] = p[k][0]; pos[k][1] = p[k][1];
    x[k][0] = p[k][0]; x[k][1] = p[k][1]; x[k][2] = 0.f; x[k][3] = 0.f;
#pragma unroll
    for (int m = 0; m < 3; ++m) {
      x[k][4 + 2 * m] = p[m][0] - p[k][0];
      x[k][5 + 2 * m] = p[m][1] - p[k][1];
    }
    const int o0 = (k == 0) ? 1 : 0;        // OTHER_IDX rows {1,2},{0,2},{0,1}
    const int o1 = (k == 2) ? 1 : 2;
    x[k][10] = p[o0][0] - p[k][0]; x[k][11] = p[o0][1] - p[k][1];
    x[k][12] = p[o1][0] - p[k][0]; x[k][13] = p[o1][1] - p[k][1];
  }
  float hO[16];
#pragma unroll
  for (int c = 0; c < 16; ++c) hO[c] = 0.f;
  gatv2_accum(x, pos, sWl, sbl, sWr, sbr, sWe, satt, sbias, hO);

  // h = [agent_pool(16), obj_pool(16)]  -> LDS, padded stride
  {
    float* hrow = hbuf + (wave * 32 + lane) * HSTR;
#pragma unroll
    for (int c = 0; c < 16; ++c) { hrow[c] = hA[c]; hrow[16 + c] = hO[c]; }
  }
  __syncthreads();

  // ================= MLP via WMMA (wave handles 2 M-tiles of 16 samples) =================
  const int g = lane >> 4;        // lane half -> K striping group
  const int m = lane & 15;        // row (A) / column (B,D) within tile
  float* zrow = zbuf + wave * 16 * ZSTR;

#pragma unroll 1
  for (int t = 0; t < 2; ++t) {
    // --- load A1 = h tile [16 x 32] as f16 hi/lo split ---
    v16h a1h, a1l;
    {
      const float* hsrc = hbuf + (wave * 32 + t * 16 + m) * HSTR;
#pragma unroll
      for (int e = 0; e < 16; ++e) {
        float v = hsrc[kmap(e, g)];
        _Float16 hi = (_Float16)v;
        a1h[e] = hi;
        a1l[e] = (_Float16)(v - (float)hi);
      }
    }
    // --- layer 1: z = relu(h @ W1 + b1), 8 column tiles of 16 ---
    // unroll 1: keep only one B tile + one accumulator live (no spills)
#pragma unroll 1
    for (int nb = 0; nb < 8; ++nb) {
      v16h bh, blo;
#pragma unroll
      for (int e = 0; e < 16; ++e) {
        float v = W1[kmap(e, g) * 128 + nb * 16 + m];
        _Float16 hi = (_Float16)v;
        bh[e]  = hi;
        blo[e] = (_Float16)(v - (float)hi);
      }
      v8f c = {};
      c = wmma_split3(a1h, a1l, bh, blo, c);
      const float bn = sb1[nb * 16 + m];
#pragma unroll
      for (int r = 0; r < 8; ++r) {
        float f = fmaxf(c[r] + bn, 0.f);
        zrow[(r + 8 * g) * ZSTR + nb * 16 + m] = f;   // z[M][n]
      }
    }
    __syncthreads();   // uniform across block; publishes z tile

    // --- layer 2: out = z @ W2 + b2  (K = 128 in 4 chunks, 2 column tiles) ---
    v8f acc[2] = {};
#pragma unroll 1
    for (int kc = 0; kc < 4; ++kc) {
      v16h ah, al;
#pragma unroll
      for (int e = 0; e < 16; ++e) {
        float v = zrow[m * ZSTR + kc * 32 + kmap(e, g)];
        _Float16 hi = (_Float16)v;
        ah[e] = hi;
        al[e] = (_Float16)(v - (float)hi);
      }
#pragma unroll
      for (int nt = 0; nt < 2; ++nt) {
        v16h bh, blo;
#pragma unroll
        for (int e = 0; e < 16; ++e) {
          float v = W2[(kc * 32 + kmap(e, g)) * 32 + nt * 16 + m];
          _Float16 hi = (_Float16)v;
          bh[e]  = hi;
          blo[e] = (_Float16)(v - (float)hi);
        }
        acc[nt] = wmma_split3(ah, al, bh, blo, acc[nt]);
      }
    }
    // --- epilogue: bias + store (no final relu) ---
#pragma unroll
    for (int nt = 0; nt < 2; ++nt) {
      const int  n2 = nt * 16 + m;
      const float bn = sb2[n2];
#pragma unroll
      for (int r = 0; r < 8; ++r) {
        const int S = gw * 32 + t * 16 + r + 8 * g;    // sample index of row M
        out[S * 32 + n2] = acc[nt][r] + bn;
      }
    }
    __syncthreads();   // z buffer reused by next t
  }
}

extern "C" void kernel_launch(void* const* d_in, const int* in_sizes, int n_in,
                              void* d_out, int out_size, void* d_ws, size_t ws_size,
                              hipStream_t stream) {
  const float* agent_pos = (const float*)d_in[0];
  const float* agent_vel = (const float*)d_in[1];
  const float* rel_lm    = (const float*)d_in[2];
  const float* other_pos = (const float*)d_in[3];
  const float* lm_pos    = (const float*)d_in[4];
  const float* Wl    = (const float*)d_in[5];
  const float* bl    = (const float*)d_in[6];
  const float* Wr    = (const float*)d_in[7];
  const float* br    = (const float*)d_in[8];
  const float* We    = (const float*)d_in[9];
  const float* att   = (const float*)d_in[10];
  const float* bias0 = (const float*)d_in[11];
  const float* W1 = (const float*)d_in[12];
  const float* b1 = (const float*)d_in[13];
  const float* W2 = (const float*)d_in[14];
  const float* b2 = (const float*)d_in[15];
  float* out = (float*)d_out;

  const int B = in_sizes[0] / 6;              // agent_pos is [B,3,2]
  const int samples_per_block = WAVES * 32;   // 128
  const int grid = B / samples_per_block;     // 131072/128 = 1024 (exact)

  scl_fused_kernel<<<grid, BLOCK, 0, stream>>>(
      agent_pos, agent_vel, rel_lm, other_pos, lm_pos,
      Wl, bl, Wr, br, We, att, bias0, W1, b1, W2, b2, out);
}
